// Net_78365973283493
// MI455X (gfx1250) — matrix operand order repaired and checked by reading, not compiled, not run on hardware
//
#include <hip/hip_runtime.h>

// ---------------------------------------------------------------------------
// logits[B,V] = gather(w_enc.T, x) @ w_dec^T     (B=4096, V=50000, K=300)
//
// MI455X (gfx1250) strategy:
//   * Output (819 MB fp32) is the HBM bottleneck (~35us floor @ 23.3 TB/s);
//     bf16 WMMA (v_wmma_f32_16x16x32_bf16, f32 accum) puts compute near the
//     same point -> balanced. fp32 WMMA (16x16x4) would be ~8x compute-bound.
//   * prep kernels write K-padded (300->320, zero-filled) bf16 copies of the
//     gathered H (2.6 MB) and w_dec (32 MB) into d_ws; w_dec bf16 stays L2
//     resident (192 MB L2) across its ~32 reuse passes.
//   * GEMM: 128x128x32 tiles, 8 waves, double-buffered LDS fed by
//     GLOBAL_LOAD_ASYNC_TO_LDS_B128 (ASYNCcnt pipeline), 80B padded LDS rows
//     (bank-conflict-free 16B fragment reads), NT stores for the streaming
//     output so it never evicts w_dec from L2.
//
// Workspace use: 4096*320*2 + 50000*320*2 = 34,621,440 bytes.
// ---------------------------------------------------------------------------

typedef __attribute__((ext_vector_type(16))) __bf16 v16bf;
typedef __attribute__((ext_vector_type(8)))  __bf16 v8bf;
typedef __attribute__((ext_vector_type(8)))  float  v8f;
typedef __attribute__((ext_vector_type(4)))  int    v4i;

// builtin expects int4 pointers: AS1 (global) source, AS3 (LDS) destination
typedef __attribute__((address_space(1))) v4i g_v4i;
typedef __attribute__((address_space(3))) v4i l_v4i;

#define VOCAB 50000
#define DIM   300
#define BATCH 4096
#define KP    320          // K padded to multiple of BK
#define BM    128
#define BN    128
#define BK    32
#define NK    (KP / BK)    // 10 K-chunks
#define LDK   40           // LDS row stride in bf16 elems (80B -> conflict-free)

// 16B global -> LDS copy: async (ASYNCcnt) when available, sync fallback.
static __device__ __forceinline__ void cp16_g2l(const __bf16* g, __bf16* l) {
#if __has_builtin(__builtin_amdgcn_global_load_async_to_lds_b128)
    __builtin_amdgcn_global_load_async_to_lds_b128((g_v4i*)g, (l_v4i*)l, 0, 0);
#else
    *reinterpret_cast<uint4*>(l) = *reinterpret_cast<const uint4*>(g);
#endif
}

static __device__ __forceinline__ void wait_async0() {
    asm volatile("s_wait_asynccnt 0x0" ::: "memory");
}

// ---------------- prep kernel 1: H[b,k] = w_enc[k, x[b]]  -> bf16, K-padded
__global__ __launch_bounds__(256)
void gather_h_bf16(const float* __restrict__ w_enc, const int* __restrict__ x,
                   unsigned short* __restrict__ Hb_raw) {
    __bf16* Hb = reinterpret_cast<__bf16*>(Hb_raw);
    int i = blockIdx.x * 256 + threadIdx.x;      // over BATCH*KP = 1,310,720
    int b = i / KP;
    int k = i - b * KP;
    float v = 0.0f;
    if (k < DIM) v = w_enc[(size_t)k * VOCAB + x[b]];
    Hb[i] = (__bf16)v;
}

// ---------------- prep kernel 2: w_dec fp32 -> bf16, K-padded
__global__ __launch_bounds__(256)
void cvt_wdec_bf16(const float* __restrict__ w_dec,
                   unsigned short* __restrict__ Wb_raw) {
    __bf16* Wb = reinterpret_cast<__bf16*>(Wb_raw);
    int i = blockIdx.x * 256 + threadIdx.x;      // over VOCAB*KP = 16,000,000
    int v = i / KP;
    int k = i - v * KP;
    float f = 0.0f;
    if (k < DIM) f = w_dec[(size_t)v * DIM + k];
    Wb[i] = (__bf16)f;
}

// ---------------- main GEMM: out[B,V] = Hb @ Wb^T  (bf16 inputs, f32 accum)
__global__ __launch_bounds__(256)
void gemm_bf16_wmma(const unsigned short* __restrict__ Hb_raw,
                    const unsigned short* __restrict__ Wb_raw,
                    float* __restrict__ out) {
    const __bf16* Hb = reinterpret_cast<const __bf16*>(Hb_raw);
    const __bf16* Wb = reinterpret_cast<const __bf16*>(Wb_raw);

    __shared__ __align__(16) __bf16 As[2][BM * LDK];
    __shared__ __align__(16) __bf16 Bs[2][BN * LDK];

    const int tid  = threadIdx.x;
    const int lane = tid & 31;
    const int wave = tid >> 5;          // 0..7
    const int wm   = wave & 3;          // 4 waves along M (32 rows each)
    const int wn   = wave >> 2;         // 2 waves along N (64 cols each)
    const int l16  = lane & 15;
    const int half = (lane >> 4) & 1;   // 0: lanes 0-15, 1: lanes 16-31

    const int m_blk = blockIdx.y * BM;  // BATCH divisible by BM
    const int n_blk = blockIdx.x * BN;  // last N tile partial (store-guarded)

    // Per-thread staging tasks: 512 16B segments per tile, 2 per thread.
    // OOB vocab rows are clamped (garbage columns are never stored).
    auto issue_copies = [&](int kc, int buf) {
        #pragma unroll
        for (int it = 0; it < 2; ++it) {
            int task = tid + it * 256;
            int row  = task >> 2;
            int seg  = task & 3;
            // A tile
            cp16_g2l(Hb + (size_t)(m_blk + row) * KP + kc * BK + seg * 8,
                     &As[buf][row * LDK + seg * 8]);
            // B tile (clamped N edge)
            int gn = n_blk + row;
            if (gn >= VOCAB) gn = VOCAB - 1;
            cp16_g2l(Wb + (size_t)gn * KP + kc * BK + seg * 8,
                     &Bs[buf][row * LDK + seg * 8]);
        }
    };

    v8f c[2][4] = {};

    issue_copies(0, 0);                 // prologue: fill buffer 0

    for (int kc = 0; kc < NK; ++kc) {
        const int buf = kc & 1;

        wait_async0();                  // own copies for chunk kc complete
        __syncthreads();                // all waves' copies visible; all waves
                                        // done reading buf^1 (compute kc-1)

        if (kc + 1 < NK)                // overlap next copy with this compute
            issue_copies(kc + 1, buf ^ 1);

        // ---- fragments per gfx1250 WMMA VGPR layouts
        // A 16x32 (16-bit): lanes 0-15 -> M=lane, K segs {0..7},{16..23}
        //                   lanes 16-31 -> M=lane-16, K segs {8..15},{24..31}
        v16bf a[2];
        #pragma unroll
        for (int mi = 0; mi < 2; ++mi) {
            int row = wm * 32 + mi * 16 + l16;
            int kb  = half * 8;
            union { v8bf h[2]; v16bf v; } u;
            u.h[0] = *reinterpret_cast<const v8bf*>(&As[buf][row * LDK + kb]);
            u.h[1] = *reinterpret_cast<const v8bf*>(&As[buf][row * LDK + kb + 16]);
            a[mi] = u.v;
        }
        // B 32x16 (16-bit): lane -> N=lane%16; lanes 0-15 hold K {0..7},{16..23},
        //                   lanes 16-31 hold K {8..15},{24..31} (same as A)
        v16bf b[4];
        #pragma unroll
        for (int ni = 0; ni < 4; ++ni) {
            int col = wn * 64 + ni * 16 + l16;
            int kb  = half * 8;
            union { v8bf h[2]; v16bf v; } u;
            u.h[0] = *reinterpret_cast<const v8bf*>(&Bs[buf][col * LDK + kb]);
            u.h[1] = *reinterpret_cast<const v8bf*>(&Bs[buf][col * LDK + kb + 16]);
            b[ni] = u.v;
        }

        // ---- 8 WMMAs per wave per K chunk
        #pragma unroll
        for (int mi = 0; mi < 2; ++mi)
            #pragma unroll
            for (int ni = 0; ni < 4; ++ni)
                c[mi][ni] = __builtin_amdgcn_wmma_f32_16x16x32_bf16(
                    false, a[mi], false, b[ni], (short)0, c[mi][ni], false, false);
    }

    // ---- epilogue: D layout: VGPR j -> M = j (+8 for lanes 16-31), N = lane%16
    // NT stores: streaming output must not evict L2-resident w_dec.
    #pragma unroll
    for (int mi = 0; mi < 2; ++mi) {
        #pragma unroll
        for (int ni = 0; ni < 4; ++ni) {
            int gm0 = m_blk + wm * 32 + mi * 16 + half * 8;
            int gn  = n_blk + wn * 64 + ni * 16 + l16;
            if (gn < VOCAB) {
                #pragma unroll
                for (int j = 0; j < 8; ++j)
                    __builtin_nontemporal_store(
                        c[mi][ni][j], &out[(size_t)(gm0 + j) * VOCAB + gn]);
            }
        }
    }
}

// ---------------------------------------------------------------------------
extern "C" void kernel_launch(void* const* d_in, const int* in_sizes, int n_in,
                              void* d_out, int out_size, void* d_ws, size_t ws_size,
                              hipStream_t stream) {
    (void)in_sizes; (void)n_in; (void)out_size; (void)ws_size;

    const int*   x     = (const int*)  d_in[0];   // [4096] int32
    const float* w_enc = (const float*)d_in[1];   // [300, 50000]
    const float* w_dec = (const float*)d_in[2];   // [50000, 300]
    float*       out   = (float*)d_out;           // [4096, 50000]

    // workspace layout (requires ~34.6 MB)
    unsigned short* Hb = (unsigned short*)d_ws;                       // 4096*320 bf16
    unsigned short* Wb = (unsigned short*)((char*)d_ws +
                                           (size_t)BATCH * KP * 2);   // 50000*320 bf16

    // prep
    gather_h_bf16<<<(BATCH * KP) / 256, 256, 0, stream>>>(w_enc, x, Hb);
    cvt_wdec_bf16<<<(VOCAB * KP) / 256, 256, 0, stream>>>(w_dec, Wb);

    // GEMM: grid covers [V=50000 (x), B=4096 (y)] in 128x128 tiles
    dim3 grid((VOCAB + BN - 1) / BN, BATCH / BM);
    gemm_bf16_wmma<<<grid, 256, 0, stream>>>(Hb, Wb, out);
}
